// PY_SYN_METAAS_38946763440397
// MI455X (gfx1250) — matrix operand order repaired and checked by reading, not compiled, not run on hardware
//
#include <hip/hip_runtime.h>

typedef __attribute__((ext_vector_type(16))) __bf16 v16bf;
typedef __attribute__((ext_vector_type(8)))  __bf16 v8bf;
typedef __attribute__((ext_vector_type(4)))  __bf16 v4bf;
typedef __attribute__((ext_vector_type(8)))  float  v8f;
typedef __attribute__((ext_vector_type(4)))  float  v4f;

#define LEAKY 0.2f

// ---------------------------------------------------------------------------
// Elementwise f32 -> bf16 (4 elements/thread, b128 load / b64 store)
__global__ __launch_bounds__(256)
void cvt_bf16(const float* __restrict__ in, __bf16* __restrict__ op, int n4) {
    int i = blockIdx.x * blockDim.x + threadIdx.x;
    if (i >= n4) return;
    v4f v = ((const v4f*)in)[i];
    ((v4bf*)op)[i] = __builtin_convertvector(v, v4bf);
}

// ---------------------------------------------------------------------------
// Layer 1 GEMM: K=4 is too small for WMMA; one thread per output element.
__global__ __launch_bounds__(256)
void gemm_k4(const float* __restrict__ X, const float* __restrict__ W,
             float* __restrict__ XW, int nrows) {
    int idx = blockIdx.x * blockDim.x + threadIdx.x;
    if (idx >= nrows * 64) return;
    int n = idx >> 6, o = idx & 63;
    const float* xr = X + n * 4;
    const float* wr = W + o * 4;
    XW[idx] = xr[0]*wr[0] + xr[1]*wr[1] + xr[2]*wr[2] + xr[3]*wr[3];
}

// ---------------------------------------------------------------------------
// WMMA GEMM on pre-converted bf16: xw = x @ W^T.
// One wave computes a 16(M) x 64(N) strip: 4 accumulators sharing one A
// operand per K-step; all tile loads are direct b128 bf16 loads.
// A operand as loaded here (contiguous k, not ISA table order): lane m=lane&15, hi=lane>>4;
//   VGPRs 0-7 hold K in [hi*16, hi*16+16)
// B operand as loaded here (same contiguous k as A): lane n=lane&15; VGPRs 0-7 hold K in [hi*16, hi*16+16)
// D-layout: lane col n=lane&15; VGPR v is row M = v + 8*hi
template<int K, int OUT>
__global__ __launch_bounds__(256)
void gat_gemm_wmma(const __bf16* __restrict__ X, const __bf16* __restrict__ W,
                   float* __restrict__ XW, int nrows) {
    const int lane = threadIdx.x & 31;
    const int wave = (blockIdx.x * blockDim.x + threadIdx.x) >> 5;
    const int nstrips = OUT / 64;
    const int mt = wave / nstrips;
    const int st = wave % nstrips;
    if (mt * 16 >= nrows) return;
    const int mrow  = mt * 16;
    const int ncol0 = st * 64;
    const int loc = lane & 15;
    const int hi  = lane >> 4;

    const __bf16* __restrict__ xrow = X + (size_t)(mrow + loc) * K;
    const __bf16* __restrict__ w0 = W + (size_t)(ncol0 +  0 + loc) * K + (hi << 4);
    const __bf16* __restrict__ w1 = W + (size_t)(ncol0 + 16 + loc) * K + (hi << 4);
    const __bf16* __restrict__ w2 = W + (size_t)(ncol0 + 32 + loc) * K + (hi << 4);
    const __bf16* __restrict__ w3 = W + (size_t)(ncol0 + 48 + loc) * K + (hi << 4);

    v8f c0 = {}, c1 = {}, c2 = {}, c3 = {};
#pragma unroll
    for (int k0 = 0; k0 < K; k0 += 32) {
        v8bf a0 = *(const v8bf*)(xrow + k0 + (hi << 4));
        v8bf a1 = *(const v8bf*)(xrow + k0 + (hi << 4) + 8);
        v16bf a = __builtin_shufflevector(a0, a1, 0,1,2,3,4,5,6,7,8,9,10,11,12,13,14,15);

        v8bf b0l = *(const v8bf*)(w0 + k0), b0h = *(const v8bf*)(w0 + k0 + 8);
        v8bf b1l = *(const v8bf*)(w1 + k0), b1h = *(const v8bf*)(w1 + k0 + 8);
        v8bf b2l = *(const v8bf*)(w2 + k0), b2h = *(const v8bf*)(w2 + k0 + 8);
        v8bf b3l = *(const v8bf*)(w3 + k0), b3h = *(const v8bf*)(w3 + k0 + 8);
        v16bf b0 = __builtin_shufflevector(b0l, b0h, 0,1,2,3,4,5,6,7,8,9,10,11,12,13,14,15);
        v16bf b1 = __builtin_shufflevector(b1l, b1h, 0,1,2,3,4,5,6,7,8,9,10,11,12,13,14,15);
        v16bf b2 = __builtin_shufflevector(b2l, b2h, 0,1,2,3,4,5,6,7,8,9,10,11,12,13,14,15);
        v16bf b3 = __builtin_shufflevector(b3l, b3h, 0,1,2,3,4,5,6,7,8,9,10,11,12,13,14,15);

        c0 = __builtin_amdgcn_wmma_f32_16x16x32_bf16(false, a, false, b0, (short)0, c0, false, false);
        c1 = __builtin_amdgcn_wmma_f32_16x16x32_bf16(false, a, false, b1, (short)0, c1, false, false);
        c2 = __builtin_amdgcn_wmma_f32_16x16x32_bf16(false, a, false, b2, (short)0, c2, false, false);
        c3 = __builtin_amdgcn_wmma_f32_16x16x32_bf16(false, a, false, b3, (short)0, c3, false, false);
    }
#pragma unroll
    for (int v = 0; v < 8; ++v) {
        float* __restrict__ dst = XW + (size_t)(mrow + v + 8*hi) * OUT + ncol0 + loc;
        dst[0]  = c0[v];
        dst[16] = c1[v];
        dst[32] = c2[v];
        dst[48] = c3[v];
    }
}

// ---------------------------------------------------------------------------
// Per-node attention scores: s_src[n] = xw[n,:]·a_src ; s_dst[n] = xw[n,:]·a_dst
template<int OUT>
__global__ __launch_bounds__(256)
void node_scores(const float* __restrict__ XW,
                 const float* __restrict__ asrc, const float* __restrict__ adst,
                 float* __restrict__ ssrc, float* __restrict__ sdst, int nrows) {
    const int wave = (blockIdx.x * blockDim.x + threadIdx.x) >> 5;
    const int lane = threadIdx.x & 31;
    if (wave >= nrows) return;
    const float* r = XW + (size_t)wave * OUT;
    float s1 = 0.f, s2 = 0.f;
#pragma unroll
    for (int o = lane; o < OUT; o += 32) {
        float v = r[o];
        s1 += v * asrc[o];
        s2 += v * adst[o];
    }
#pragma unroll
    for (int off = 16; off; off >>= 1) {
        s1 += __shfl_xor(s1, off, 32);
        s2 += __shfl_xor(s2, off, 32);
    }
    if (lane == 0) { ssrc[wave] = s1; sdst[wave] = s2; }
}

// ---------------------------------------------------------------------------
// Init: m=-inf, ssum=0, out[n,o]=bias[o]
template<int OUT>
__global__ __launch_bounds__(256)
void gat_init(float* __restrict__ m, float* __restrict__ ssum,
              float* __restrict__ out, const float* __restrict__ bias, int nrows) {
    int idx = blockIdx.x * blockDim.x + threadIdx.x;
    if (idx < nrows) { m[idx] = -__builtin_inff(); ssum[idx] = 0.f; }
    if (idx < nrows * OUT) out[idx] = bias[idx & (OUT - 1)];
}

// ordered-float atomic max (init must be -inf)
__device__ __forceinline__ void atomicMaxF(float* addr, float v) {
    if (v >= 0.f) atomicMax((int*)addr, __float_as_int(v));
    else          atomicMin((unsigned int*)addr, __float_as_uint(v));
}

// ---------------------------------------------------------------------------
// Edge pass 1: e = leaky(s_src[src]+s_dst[dst]); segment max into m[dst]
__global__ __launch_bounds__(256)
void edge_max(const int* __restrict__ row, const int* __restrict__ col,
              const float* __restrict__ ssrc, const float* __restrict__ sdst,
              float* __restrict__ emsg, float* __restrict__ m, int E, int Nn) {
    int k = blockIdx.x * blockDim.x + threadIdx.x;
    if (k >= E + Nn) return;
    int s, d;
    if (k < E) { s = row[k]; d = col[k]; } else { s = d = k - E; }
    float e = ssrc[s] + sdst[d];
    e = e > 0.f ? e : LEAKY * e;
    emsg[k] = e;
    atomicMaxF(m + d, e);
}

// Edge pass 2: ex = exp(e - m[dst]); segment sum into ssum[dst]; emsg <- ex
__global__ __launch_bounds__(256)
void edge_expsum(const int* __restrict__ col, const float* __restrict__ m,
                 float* __restrict__ emsg, float* __restrict__ ssum, int E, int Nn) {
    int k = blockIdx.x * blockDim.x + threadIdx.x;
    if (k >= E + Nn) return;
    int d = (k < E) ? col[k] : k - E;
    float ex = __expf(emsg[k] - m[d]);
    emsg[k] = ex;
    atomicAdd(ssum + d, ex);
}

// Edge pass 3: alpha = ex/ssum[dst]; out[dst,:] += alpha * xw[src,:]
// Grid-stride waves; prefetch next message's source row to hide L2 latency.
template<int OUT>
__global__ __launch_bounds__(256)
void edge_scatter(const int* __restrict__ row, const int* __restrict__ col,
                  const float* __restrict__ emsg, const float* __restrict__ ssum,
                  const float* __restrict__ XW, float* __restrict__ out, int E, int Nn) {
    const int lane   = threadIdx.x & 31;
    const int gwave  = (blockIdx.x * blockDim.x + threadIdx.x) >> 5;
    const int nwaves = (gridDim.x * blockDim.x) >> 5;
    const int M = E + Nn;
    for (int k = gwave; k < M; k += nwaves) {
        const int k2 = k + nwaves;
        if (k2 < M) {  // prefetch next iteration's gather row (OUT/8 bytes per lane)
            const int s2 = (k2 < E) ? row[k2] : k2 - E;
            __builtin_prefetch(XW + (size_t)s2 * OUT + (size_t)lane * (OUT / 32), 0, 0);
        }
        int s, d;
        if (k < E) { s = row[k]; d = col[k]; } else { s = d = k - E; }
        const float alpha = emsg[k] / ssum[d];
        const float* __restrict__ src = XW + (size_t)s * OUT;
        float* __restrict__ dst = out + (size_t)d * OUT;
#pragma unroll
        for (int o = lane; o < OUT; o += 32)
            atomicAdd(dst + o, alpha * src[o]);
    }
}

// ---------------------------------------------------------------------------
// EdgeModel: relu(We[0:512]·h[row] + We[512:1024]·h[col] + We[1024]*ea + be)
// Grid-stride waves with prefetch of next edge's two h-rows.
__global__ __launch_bounds__(256)
void edge_model(const int* __restrict__ row, const int* __restrict__ col,
                const float* __restrict__ h, const float* __restrict__ eattr,
                const float* __restrict__ We, const float* __restrict__ be,
                float* __restrict__ out, int E) {
    const int lane   = threadIdx.x & 31;
    const int gwave  = (blockIdx.x * blockDim.x + threadIdx.x) >> 5;
    const int nwaves = (gridDim.x * blockDim.x) >> 5;
    for (int e = gwave; e < E; e += nwaves) {
        const int e2 = e + nwaves;
        if (e2 < E) {  // 32 lanes x 64B = full 2KB row each
            __builtin_prefetch(h + (size_t)row[e2] * 512 + lane * 16, 0, 0);
            __builtin_prefetch(h + (size_t)col[e2] * 512 + lane * 16, 0, 0);
        }
        const float* __restrict__ hs = h + (size_t)row[e] * 512;
        const float* __restrict__ hd = h + (size_t)col[e] * 512;
        float acc = 0.f;
#pragma unroll 4
        for (int j = lane; j < 512; j += 32)
            acc += We[j] * hs[j] + We[512 + j] * hd[j];
#pragma unroll
        for (int off = 16; off; off >>= 1)
            acc += __shfl_xor(acc, off, 32);
        if (lane == 0) {
            float v = acc + We[1024] * eattr[e] + be[0];
            out[e] = v > 0.f ? v : 0.f;
        }
    }
}

// ---------------------------------------------------------------------------
extern "C" void kernel_launch(void* const* d_in, const int* in_sizes, int n_in,
                              void* d_out, int out_size, void* d_ws, size_t ws_size,
                              hipStream_t stream) {
    const int N = in_sizes[0] / 4;       // 20000
    const int E = in_sizes[1] / 2;       // 160000
    const int M = E + N;                 // messages incl. self-loops

    const float* x     = (const float*)d_in[0];
    const int*   ei    = (const int*)d_in[1];
    const int*   row   = ei;
    const int*   col   = ei + E;
    const float* eattr = (const float*)d_in[2];
    const float* W1 = (const float*)d_in[3];
    const float* as1 = (const float*)d_in[4];
    const float* ad1 = (const float*)d_in[5];
    const float* b1  = (const float*)d_in[6];
    const float* W2 = (const float*)d_in[7];
    const float* as2 = (const float*)d_in[8];
    const float* ad2 = (const float*)d_in[9];
    const float* b2  = (const float*)d_in[10];
    const float* W3 = (const float*)d_in[11];
    const float* as3 = (const float*)d_in[12];
    const float* ad3 = (const float*)d_in[13];
    const float* b3  = (const float*)d_in[14];
    const float* We  = (const float*)d_in[15];
    const float* be  = (const float*)d_in[16];
    float* out = (float*)d_out;

    // ---- workspace partition (256B aligned) ----
    char* ws = (char*)d_ws;
    size_t off = 0;
    auto alloc = [&](size_t bytes) -> void* {
        void* p = (void*)(ws + off);
        off = (off + bytes + 255) & ~(size_t)255;
        return p;
    };
    float*  bufXW  = (float*)alloc((size_t)N * 512 * 4);   // xw of current layer
    float*  bufOut = (float*)alloc((size_t)N * 512 * 4);   // aggregated output
    __bf16* xbf    = (__bf16*)alloc((size_t)N * 512 * 2);  // bf16 copy of GEMM input
    __bf16* wbf    = (__bf16*)alloc((size_t)512 * 128 * 2);// bf16 copy of weights
    float*  ssrc   = (float*)alloc((size_t)N * 4);
    float*  sdst   = (float*)alloc((size_t)N * 4);
    float*  mmax   = (float*)alloc((size_t)N * 4);
    float*  ssum   = (float*)alloc((size_t)N * 4);
    float*  emsg   = (float*)alloc((size_t)M * 4);
    (void)ws_size;

    const int T = 256;
    const int GS_BLOCKS = 4096;          // grid-stride kernels: 32768 waves
    auto cdiv = [](int a, int b) { return (a + b - 1) / b; };

    // =================== Layer 1: 4 -> 64 ===================
    {
        const int OUT = 64;
        gemm_k4<<<cdiv(N * OUT, T), T, 0, stream>>>(x, W1, bufXW, N);
        node_scores<64><<<cdiv(N * 32, T), T, 0, stream>>>(bufXW, as1, ad1, ssrc, sdst, N);
        gat_init<64><<<cdiv(N * OUT, T), T, 0, stream>>>(mmax, ssum, bufOut, b1, N);
        edge_max<<<cdiv(M, T), T, 0, stream>>>(row, col, ssrc, sdst, emsg, mmax, E, N);
        edge_expsum<<<cdiv(M, T), T, 0, stream>>>(col, mmax, emsg, ssum, E, N);
        edge_scatter<64><<<GS_BLOCKS, T, 0, stream>>>(row, col, emsg, ssum, bufXW, bufOut, E, N);
    }
    // =================== Layer 2: 64 -> 128 ===================
    {
        const int K = 64, OUT = 128;
        cvt_bf16<<<cdiv(N * K / 4, T), T, 0, stream>>>(bufOut, xbf, N * K / 4);
        cvt_bf16<<<cdiv(OUT * K / 4, T), T, 0, stream>>>(W2, wbf, OUT * K / 4);
        const int waves = (N / 16) * (OUT / 64);
        gat_gemm_wmma<64, 128><<<cdiv(waves * 32, T), T, 0, stream>>>(xbf, wbf, bufXW, N);
        node_scores<128><<<cdiv(N * 32, T), T, 0, stream>>>(bufXW, as2, ad2, ssrc, sdst, N);
        gat_init<128><<<cdiv(N * OUT, T), T, 0, stream>>>(mmax, ssum, bufOut, b2, N);
        edge_max<<<cdiv(M, T), T, 0, stream>>>(row, col, ssrc, sdst, emsg, mmax, E, N);
        edge_expsum<<<cdiv(M, T), T, 0, stream>>>(col, mmax, emsg, ssum, E, N);
        edge_scatter<128><<<GS_BLOCKS, T, 0, stream>>>(row, col, emsg, ssum, bufXW, bufOut, E, N);
    }
    // =================== Layer 3: 128 -> 512 ===================
    {
        const int K = 128, OUT = 512;
        cvt_bf16<<<cdiv(N * K / 4, T), T, 0, stream>>>(bufOut, xbf, N * K / 4);
        cvt_bf16<<<cdiv(OUT * K / 4, T), T, 0, stream>>>(W3, wbf, OUT * K / 4);
        const int waves = (N / 16) * (OUT / 64);
        gat_gemm_wmma<128, 512><<<cdiv(waves * 32, T), T, 0, stream>>>(xbf, wbf, bufXW, N);
        node_scores<512><<<cdiv(N * 32, T), T, 0, stream>>>(bufXW, as3, ad3, ssrc, sdst, N);
        gat_init<512><<<cdiv(N * OUT, T), T, 0, stream>>>(mmax, ssum, bufOut, b3, N);
        edge_max<<<cdiv(M, T), T, 0, stream>>>(row, col, ssrc, sdst, emsg, mmax, E, N);
        edge_expsum<<<cdiv(M, T), T, 0, stream>>>(col, mmax, emsg, ssum, E, N);
        edge_scatter<512><<<GS_BLOCKS, T, 0, stream>>>(row, col, emsg, ssum, bufXW, bufOut, E, N);
    }
    // =================== EdgeModel (the output) ===================
    edge_model<<<GS_BLOCKS, T, 0, stream>>>(row, col, bufOut, eattr, We, be, out, E);
}
